// RNNModel_7593502179384
// MI455X (gfx1250) — matrix operand order repaired and checked by reading, not compiled, not run on hardware
//
#include <hip/hip_runtime.h>
#include <hip/hip_bf16.h>

typedef __attribute__((ext_vector_type(16))) __bf16 v16bf;
typedef __attribute__((ext_vector_type(8)))  __bf16 v8bf;
typedef __attribute__((ext_vector_type(8)))  float  v8f;
typedef __attribute__((ext_vector_type(4)))  float  v4f;

#if __has_builtin(__builtin_amdgcn_tanhf)
#define TANHF(x) __builtin_amdgcn_tanhf(x)
#elif __has_builtin(__builtin_amdgcn_tanh_f32)
#define TANHF(x) __builtin_amdgcn_tanh_f32(x)
#else
#define TANHF(x) tanhf(x)
#endif

#define BB 256
#define TT 512
#define II 128
#define HH 256
#define LDH 264   // h row stride (bf16): 528B = 132 dwords -> bank-offset 4/row, conflict-free b128 halves
#define LDX 136   // staged x row stride (bf16): 272B = 68 dwords -> bank-offset 4/row

// D = A*B + C, 16x16x32 bf16 -> f32
static __device__ __forceinline__ v8f wmma_bf16(v16bf a, v16bf b, v8f c) {
  return __builtin_amdgcn_wmma_f32_16x16x32_bf16(false, a, false, b, (short)0, c, false, false);
}

static __device__ __forceinline__ v16bf cat8(v8bf lo, v8bf hi) {
  return __builtin_shufflevector(lo, hi, 0,1,2,3,4,5,6,7,8,9,10,11,12,13,14,15);
}

// A fragment (16-bit A layout): lane half 0 -> K {kb..kb+7, kb+16..kb+23}, half 1 -> +8.
static __device__ __forceinline__ v16bf loadA(const __bf16* base) {
  return cat8(*(const v8bf*)base, *(const v8bf*)(base + 16));
}

// Load f32 p[0..7] and p[16..23] from global, convert to one bf16 B-fragment lane slice.
static __device__ __forceinline__ v16bf cvt16_global(const float* __restrict__ p) {
  v4f a = *(const v4f*)(p);
  v4f b = *(const v4f*)(p + 4);
  v4f c = *(const v4f*)(p + 16);
  v4f d = *(const v4f*)(p + 20);
  v16bf r;
  #pragma unroll
  for (int i = 0; i < 4; ++i) {
    r[i]      = (__bf16)a[i];
    r[i + 4]  = (__bf16)b[i];
    r[i + 8]  = (__bf16)c[i];
    r[i + 12] = (__bf16)d[i];
  }
  return r;
}

__global__ __launch_bounds__(256, 1)
void rnn_fused_kernel(const float* __restrict__ x,
                      const float* __restrict__ Wih,
                      const float* __restrict__ Whh,
                      const float* __restrict__ bih,
                      const float* __restrict__ bhh,
                      const float* __restrict__ fcw,
                      const float* __restrict__ fcb,
                      float* __restrict__ out)
{
  __shared__ __bf16 sh[2][16 * LDH];   // double-buffered hidden state (bf16)
  __shared__ __bf16 sx[2][16 * LDX];   // double-buffered staged x_t (bf16)
  __shared__ float  sred[16];

  const int tid  = threadIdx.x;
  const int lane = tid & 31;
  const int wave = tid >> 5;             // 0..7
  const int b0   = blockIdx.x * 16;      // batch tile base

  const int n0   = wave * 32;            // this wave's 2 column tiles: n0, n0+16
  const int ksel = lane >> 4;            // lane half selector
  const int l16  = lane & 15;

  // ---- zero h[0] ----
  for (int idx = tid; idx < 16 * LDH; idx += 256) sh[0][idx] = (__bf16)0.0f;

  // ---- per-lane bias (column N = n_tile + l16, same for all 8 acc VGPRs) ----
  const float bias0 = bih[n0 + l16]      + bhh[n0 + l16];
  const float bias1 = bih[n0 + 16 + l16] + bhh[n0 + 16 + l16];

  // ---- loop-invariant weight B-fragments -> registers (global -> cvt bf16) ----
  // B frag layout (16-bit, 32x16): lane holds col n = n_tile + l16,
  // K {kb..kb+7, kb+16..kb+23} with kb = k_base + (lane>=16 ? 8 : 0).
  v16bf wh[HH / 32][2];   // 8 kt x 2 tiles x 8 VGPRs = 128 VGPRs
  v16bf wi[II / 32][2];   // 4 kt x 2 tiles x 8 VGPRs =  64 VGPRs
  #pragma unroll
  for (int kt = 0; kt < HH / 32; ++kt) {
    #pragma unroll
    for (int tile = 0; tile < 2; ++tile) {
      const int n = n0 + tile * 16 + l16;
      wh[kt][tile] = cvt16_global(Whh + (size_t)n * HH + kt * 32 + ksel * 8);
    }
  }
  #pragma unroll
  for (int kt = 0; kt < II / 32; ++kt) {
    #pragma unroll
    for (int tile = 0; tile < 2; ++tile) {
      const int n = n0 + tile * 16 + l16;
      wi[kt][tile] = cvt16_global(Wih + (size_t)n * II + kt * 32 + ksel * 8);
    }
  }

  // ---- x staging slice for this thread: row xm, 8 floats at col xi ----
  const int xm = tid >> 4;
  const int xi = (tid & 15) * 8;
  const float* xrow = x + ((size_t)(b0 + xm) * TT) * II + xi;

  // prologue: stage x for t=0 into sx[0]
  {
    v4f f0 = *(const v4f*)(xrow);
    v4f f1 = *(const v4f*)(xrow + 4);
    v8bf xb;
    #pragma unroll
    for (int i = 0; i < 4; ++i) { xb[i] = (__bf16)f0[i]; xb[i + 4] = (__bf16)f1[i]; }
    *(v8bf*)&sx[0][xm * LDX + xi] = xb;
  }
  __syncthreads();

  int cur = 0;
  for (int t = 0; t < TT; ++t) {
    const int p = t & 1;

    // issue global loads for x(t+1) early; latency hidden behind 24 WMMAs
    v4f f0 = {0.f, 0.f, 0.f, 0.f}, f1 = {0.f, 0.f, 0.f, 0.f};
    if (t + 1 < TT) {
      const float* xp = xrow + (size_t)(t + 1) * II;
      f0 = *(const v4f*)(xp);
      f1 = *(const v4f*)(xp + 4);
      if (t + 2 < TT) __builtin_prefetch(xrow + (size_t)(t + 2) * II, 0, 3);
    }

    v8f acc0, acc1;
    #pragma unroll
    for (int i = 0; i < 8; ++i) { acc0[i] = bias0; acc1[i] = bias1; }

    const __bf16* axb = &sx[p][l16 * LDX + ksel * 8];       // x A-frag lane base
    const __bf16* ahb = &sh[cur][l16 * LDH + ksel * 8];     // h A-frag lane base

    // ---- software-pipelined A fragments: 2-deep rotating buffer so the
    //      kt+1 ds_loads overlap the kt WMMAs (partial dscnt waits, no WAR nops)
    v16bf a_cur = loadA(axb);

    // input projection: x_t (16xI) * Wih^T, B from registers
    #pragma unroll
    for (int kt = 0; kt < II / 32; ++kt) {
      v16bf a_nxt = (kt + 1 < II / 32) ? loadA(axb + (kt + 1) * 32)
                                       : loadA(ahb);          // chain into hidden GEMM
      acc0 = wmma_bf16(a_cur, wi[kt][0], acc0);
      acc1 = wmma_bf16(a_cur, wi[kt][1], acc1);
      a_cur = a_nxt;
    }

    // recurrence: h (16xH) * Whh^T, B from registers
    #pragma unroll
    for (int kt = 0; kt < HH / 32; ++kt) {
      v16bf a_nxt = (kt + 1 < HH / 32) ? loadA(ahb + (kt + 1) * 32) : a_cur;
      acc0 = wmma_bf16(a_cur, wh[kt][0], acc0);
      acc1 = wmma_bf16(a_cur, wh[kt][1], acc1);
      a_cur = a_nxt;
    }

    // ---- tanh + write new h to other buffer ----
    const int nxt = cur ^ 1;
    #pragma unroll
    for (int r = 0; r < 8; ++r) {
      // C/D layout: VGPR r -> row M = r (lanes 0-15) / r+8 (lanes 16-31); N = n_tile + l16
      const int row = r + ksel * 8;
      sh[nxt][row * LDH + n0 + l16]      = (__bf16)TANHF(acc0[r]);
      sh[nxt][row * LDH + n0 + 16 + l16] = (__bf16)TANHF(acc1[r]);
    }

    // ---- publish staged x(t+1) into the other x buffer ----
    if (t + 1 < TT) {
      v8bf xb;
      #pragma unroll
      for (int i = 0; i < 4; ++i) { xb[i] = (__bf16)f0[i]; xb[i + 4] = (__bf16)f1[i]; }
      *(v8bf*)&sx[p ^ 1][xm * LDX + xi] = xb;
    }

    // single barrier per step: publishes sh[nxt] and sx[p^1]; double-buffering
    // gives one-iteration slack on all read buffers.
    __syncthreads();
    cur = nxt;
  }

  // ---- FC head: out[b0+m] = sum_h h_last[m][h]*fcw[h] + fcb[0] ----
  if (tid < 16) sred[tid] = 0.0f;
  __syncthreads();
  {
    const int m  = tid & 15;
    const int cs = (tid >> 4) * 16;
    float pacc = 0.0f;
    #pragma unroll
    for (int c = 0; c < 16; ++c)
      pacc += (float)sh[cur][m * LDH + cs + c] * fcw[cs + c];
    atomicAdd(&sred[m], pacc);   // ds_add_f32
  }
  __syncthreads();
  if (tid < 16) out[b0 + tid] = sred[tid] + fcb[0];
}

extern "C" void kernel_launch(void* const* d_in, const int* in_sizes, int n_in,
                              void* d_out, int out_size, void* d_ws, size_t ws_size,
                              hipStream_t stream) {
  const float* x    = (const float*)d_in[0];
  const float* Wih  = (const float*)d_in[1];
  const float* Whh  = (const float*)d_in[2];
  const float* bih  = (const float*)d_in[3];
  const float* bhh  = (const float*)d_in[4];
  const float* fcw  = (const float*)d_in[5];
  const float* fcb  = (const float*)d_in[6];
  float* out = (float*)d_out;
  (void)in_sizes; (void)n_in; (void)d_ws; (void)ws_size; (void)out_size;

  rnn_fused_kernel<<<BB / 16, 256, 0, stream>>>(x, Wih, Whh, bih, bhh, fcw, fcb, out);
}
